// MyMoEModel_38104949850290
// MI455X (gfx1250) — compile-verified, matrix operand order checked by reading, not run on hardware
//
#include <hip/hip_runtime.h>
#include <hip/hip_bf16.h>

typedef __attribute__((ext_vector_type(16))) _Float16 v16h;
typedef __attribute__((ext_vector_type(8)))  _Float16 v8h;
typedef __attribute__((ext_vector_type(8)))  float    v8f;
typedef __attribute__((ext_vector_type(4)))  unsigned int u32x4;
typedef __attribute__((ext_vector_type(8)))  unsigned int u32x8;

#define S_TOK 65536
#define M_IN  784
#define E_EXP 8
#define H_HID 512
#define O_OUT 10
#define CAP   16384      // 2*S/E
#define KPAD  800        // M padded to multiple of 32
#define NBLK  256        // S_TOK / 256
#define AKCH  160        // K-chunk staged in LDS per TDM load (KPAD = 5*AKCH)

// ---------------------------------------------------------------------------
// 1. Gating logits: logits[s,e] = sum_m x[s,m]*wg[m,e]. One wave per token.
// ---------------------------------------------------------------------------
__global__ void __launch_bounds__(256)
moe_gate_logits(const float* __restrict__ x, const float* __restrict__ wg,
                float* __restrict__ logits) {
    int wid  = threadIdx.x >> 5;
    int lane = threadIdx.x & 31;
    int tok  = blockIdx.x * 8 + wid;
    const float* xr = x + (size_t)tok * M_IN;
    float acc[E_EXP];
#pragma unroll
    for (int e = 0; e < E_EXP; ++e) acc[e] = 0.f;
    for (int m = lane; m < M_IN; m += 32) {
        float xv = xr[m];
        const float* wrow = wg + m * E_EXP;
#pragma unroll
        for (int e = 0; e < E_EXP; ++e) acc[e] += xv * wrow[e];
    }
#pragma unroll
    for (int e = 0; e < E_EXP; ++e) {
#pragma unroll
        for (int off = 16; off > 0; off >>= 1)
            acc[e] += __shfl_xor(acc[e], off, 32);
    }
    if (lane == 0) {
#pragma unroll
        for (int e = 0; e < E_EXP; ++e) logits[(size_t)tok * E_EXP + e] = acc[e];
    }
}

// ---------------------------------------------------------------------------
// 2. Per-token softmax + top1/top2 + per-block expert histograms & gate sums
// ---------------------------------------------------------------------------
__global__ void __launch_bounds__(256)
moe_route(const float* __restrict__ logits,
          int* __restrict__ idx1, int* __restrict__ idx2,
          float* __restrict__ g1, float* __restrict__ g2,
          int* __restrict__ cnt1, int* __restrict__ cnt2,
          float* __restrict__ gsum) {
    __shared__ int   sc1[E_EXP];
    __shared__ int   sc2[E_EXP];
    __shared__ float sg[E_EXP];
    int t = threadIdx.x, b = blockIdx.x;
    int s = b * 256 + t;
    if (t < E_EXP) { sc1[t] = 0; sc2[t] = 0; sg[t] = 0.f; }
    __syncthreads();

    const float* lr = logits + (size_t)s * E_EXP;
    float l[E_EXP], g[E_EXP];
    float mx = -1e30f;
#pragma unroll
    for (int e = 0; e < E_EXP; ++e) { l[e] = lr[e]; mx = fmaxf(mx, l[e]); }
    float sum = 0.f;
#pragma unroll
    for (int e = 0; e < E_EXP; ++e) { g[e] = __expf(l[e] - mx); sum += g[e]; }
    float inv = 1.f / sum;
#pragma unroll
    for (int e = 0; e < E_EXP; ++e) g[e] *= inv;

    int i1 = 0; float best = g[0];
#pragma unroll
    for (int e = 1; e < E_EXP; ++e) if (g[e] > best) { best = g[e]; i1 = e; }
    int i2 = (i1 == 0) ? 1 : 0; float best2 = l[i2];
#pragma unroll
    for (int e = 0; e < E_EXP; ++e)
        if (e != i1 && l[e] > best2) { best2 = l[e]; i2 = e; }

    idx1[s] = i1; idx2[s] = i2;
    g1[s] = g[i1]; g2[s] = g[i2];     // raw top gates, normalized later
    atomicAdd(&sc1[i1], 1);
    atomicAdd(&sc2[i2], 1);
#pragma unroll
    for (int e = 0; e < E_EXP; ++e) atomicAdd(&sg[e], g[e]);
    __syncthreads();
    if (t < E_EXP) {
        cnt1[b * E_EXP + t] = sc1[t];
        cnt2[b * E_EXP + t] = sc2[t];
        gsum[b * E_EXP + t] = sg[t];
    }
}

// ---------------------------------------------------------------------------
// 3. Single-block scan over NBLK block histograms -> exclusive prefixes,
//    totals, and l_aux.
// ---------------------------------------------------------------------------
__global__ void __launch_bounds__(NBLK)
moe_scan(const int* __restrict__ cnt1, const int* __restrict__ cnt2,
         const float* __restrict__ gsum,
         int* __restrict__ pre1, int* __restrict__ pre2,
         int* __restrict__ tot1, float* __restrict__ laux_out) {
    __shared__ int   ibuf[NBLK];
    __shared__ float fbuf[NBLK];
    __shared__ int   tot1s[E_EXP];
    __shared__ float gts[E_EXP];
    int t = threadIdx.x;

    for (int e = 0; e < E_EXP; ++e) {
        // exclusive scan of cnt1[:,e]
        int v = cnt1[t * E_EXP + e];
        ibuf[t] = v; __syncthreads();
        int incl = v;
        for (int off = 1; off < NBLK; off <<= 1) {
            int add = (t >= off) ? ibuf[t - off] : 0;
            __syncthreads();
            incl += add;
            ibuf[t] = incl;
            __syncthreads();
        }
        pre1[t * E_EXP + e] = incl - v;
        if (t == NBLK - 1) tot1s[e] = incl;
        __syncthreads();

        // exclusive scan of cnt2[:,e]
        v = cnt2[t * E_EXP + e];
        ibuf[t] = v; __syncthreads();
        incl = v;
        for (int off = 1; off < NBLK; off <<= 1) {
            int add = (t >= off) ? ibuf[t - off] : 0;
            __syncthreads();
            incl += add;
            ibuf[t] = incl;
            __syncthreads();
        }
        pre2[t * E_EXP + e] = incl - v;
        __syncthreads();

        // reduce gate sums
        fbuf[t] = gsum[t * E_EXP + e];
        __syncthreads();
        for (int off = NBLK / 2; off > 0; off >>= 1) {
            if (t < off) fbuf[t] += fbuf[t + off];
            __syncthreads();
        }
        if (t == 0) gts[e] = fbuf[0];
        __syncthreads();
    }

    if (t < E_EXP) tot1[t] = tot1s[t];
    if (t == 0) {
        float acc = 0.f;
        for (int e = 0; e < E_EXP; ++e) {
            float me = gts[e] / (float)S_TOK;
            float ce = (float)tot1s[e] / (float)S_TOK;
            acc += me * ce;
        }
        laux_out[0] = acc * (float)E_EXP;  // mean(me*ce)*E^2
    }
}

// ---------------------------------------------------------------------------
// 4. Finalize routes: within-block ranks, capacity drop, gate normalization.
// ---------------------------------------------------------------------------
__global__ void __launch_bounds__(256)
moe_finalize(const int* __restrict__ idx1, const int* __restrict__ idx2,
             const int* __restrict__ pre1, const int* __restrict__ pre2,
             const int* __restrict__ tot1,
             float* __restrict__ g1, float* __restrict__ g2,
             int* __restrict__ loc1, int* __restrict__ loc2) {
    __shared__ int se1[256];
    __shared__ int se2[256];
    int t = threadIdx.x, b = blockIdx.x;
    int s = b * 256 + t;
    int i1 = idx1[s], i2 = idx2[s];
    se1[t] = i1; se2[t] = i2;
    __syncthreads();
    int r1 = 0, r2 = 0;
    for (int j = 0; j < t; ++j) {
        r1 += (se1[j] == i1);
        r2 += (se2[j] == i2);
    }
    int l1 = pre1[b * E_EXP + i1] + r1;
    int l2 = pre2[b * E_EXP + i2] + r2 + tot1[i2];
    bool k1 = l1 < CAP, k2 = l2 < CAP;
    float a  = k1 ? g1[s] : 0.f;
    float bb = k2 ? g2[s] : 0.f;
    float den = a + bb;
    const float eps = 1.1920929e-07f;
    if (den < eps) den = eps;
    g1[s] = a / den;
    g2[s] = bb / den;
    loc1[s] = k1 ? l1 : -1;
    loc2[s] = k2 ? l2 : -1;
}

// ---------------------------------------------------------------------------
// 5. Zero fill (16B granularity, grid-stride)
// ---------------------------------------------------------------------------
__global__ void moe_zero16(uint4* __restrict__ p, size_t n16) {
    size_t i = (size_t)blockIdx.x * blockDim.x + threadIdx.x;
    size_t stride = (size_t)gridDim.x * blockDim.x;
    uint4 z; z.x = 0; z.y = 0; z.z = 0; z.w = 0;
    for (; i < n16; i += stride) p[i] = z;
}

// ---------------------------------------------------------------------------
// 6/7. Weight prep: w1 [E,M,H] f32 -> w1t [E,H,KPAD] f16 (zero-padded K)
//      w2 [E,H,O] f32 -> w2t [E,16,H] f16 (zero-padded N)
// ---------------------------------------------------------------------------
__global__ void moe_prep_w1(const float* __restrict__ w1, _Float16* __restrict__ w1t) {
    int i = blockIdx.x * 256 + threadIdx.x;
    const int total = E_EXP * H_HID * KPAD;
    if (i >= total) return;
    int e = i / (H_HID * KPAD);
    int rem = i % (H_HID * KPAD);
    int h = rem / KPAD;
    int k = rem % KPAD;
    float v = (k < M_IN) ? w1[((size_t)e * M_IN + k) * H_HID + h] : 0.f;
    w1t[i] = (_Float16)v;
}

__global__ void moe_prep_w2(const float* __restrict__ w2, _Float16* __restrict__ w2t) {
    int i = blockIdx.x * 256 + threadIdx.x;
    const int total = E_EXP * 16 * H_HID;
    if (i >= total) return;
    int e = i / (16 * H_HID);
    int rem = i % (16 * H_HID);
    int n = rem / H_HID;
    int k = rem % H_HID;
    float v = (n < O_OUT) ? w2[((size_t)e * H_HID + k) * O_OUT + n] : 0.f;
    w2t[i] = (_Float16)v;
}

// ---------------------------------------------------------------------------
// 8. Dispatch scatter: block per token, f32 -> f16 into disp[E,CAP,KPAD]
// ---------------------------------------------------------------------------
__global__ void __launch_bounds__(256)
moe_dispatch(const float* __restrict__ x,
             const int* __restrict__ idx1, const int* __restrict__ idx2,
             const int* __restrict__ loc1, const int* __restrict__ loc2,
             _Float16* __restrict__ disp) {
    int s = blockIdx.x;
    int t = threadIdx.x;
    const float* xr = x + (size_t)s * M_IN;
    _Float16 v[4];
#pragma unroll
    for (int r = 0; r < 4; ++r) {
        int m = t + r * 256;
        v[r] = (m < M_IN) ? (_Float16)xr[m] : (_Float16)0.f;
    }
    int l1 = loc1[s];
    if (l1 >= 0) {
        _Float16* dst = disp + ((size_t)idx1[s] * CAP + l1) * KPAD;
#pragma unroll
        for (int r = 0; r < 4; ++r) {
            int m = t + r * 256;
            if (m < M_IN) dst[m] = v[r];
        }
    }
    int l2 = loc2[s];
    if (l2 >= 0) {
        _Float16* dst = disp + ((size_t)idx2[s] * CAP + l2) * KPAD;
#pragma unroll
        for (int r = 0; r < 4; ++r) {
            int m = t + r * 256;
            if (m < M_IN) dst[m] = v[r];
        }
    }
}

// ---------------------------------------------------------------------------
// WMMA helpers: 16-bit A/B fragment per ISA layout. Lane (0..15): kbase=0,
// lane(16..31): kbase=8; holds K in [kbase,kbase+8) and [kbase+16,kbase+24).
// Both chunks are 16B contiguous in a row-major [rows, K] f16 buffer.
// ---------------------------------------------------------------------------
__device__ __forceinline__ v16h moe_load_frag(const _Float16* p) {
    v8h lo = *(const v8h*)p;
    v8h hi = *(const v8h*)(p + 16);
    return __builtin_shufflevector(lo, hi, 0, 1, 2, 3, 4, 5, 6, 7,
                                           8, 9, 10, 11, 12, 13, 14, 15);
}

// ---------------------------------------------------------------------------
// Tensor Data Mover: 2-D tile load (AKCH x 32 rows of f16) global -> LDS.
// D# built per CDNA5 ISA ch.8: group0 = {count=1 | lds_addr | global_addr |
// type=2}, group1 = {data_size=2B, tensor_dim0=KPAD, tensor_dim1=CAP,
// tile_dim0=AKCH, tile_dim1=32, tensor_dim0_stride=KPAD}. Groups 2/3 omitted
// (2-D tensor). Tracked by TENSORcnt.
// ---------------------------------------------------------------------------
__device__ __forceinline__ void moe_tdm_load_a(unsigned lds_addr,
                                               const _Float16* gaddr) {
    unsigned long long ga = (unsigned long long)(size_t)gaddr;
    u32x4 g0;
    g0[0] = 1u;                                            // count=1, user mode
    g0[1] = lds_addr;                                      // LDS byte address
    g0[2] = (unsigned)ga;                                  // global_addr[31:0]
    g0[3] = (unsigned)((ga >> 32) & 0x01FFFFFFu) | (2u << 30); // [56:32] | type=2
    u32x8 g1;
    g1[0] = 0x00010000u;               // data_size = 1 (2 bytes), no flags
    g1[1] = ((unsigned)KPAD) << 16;    // tensor_dim0[15:0]
    g1[2] = ((unsigned)CAP) << 16;     // tensor_dim0[31:16]=0 | tensor_dim1[15:0]
    g1[3] = ((unsigned)AKCH) << 16;    // tensor_dim1[31:16]=0 | tile_dim0
    g1[4] = 32u;                       // tile_dim1=32, tile_dim2=0
    g1[5] = (unsigned)KPAD;            // tensor_dim0_stride[31:0]
    g1[6] = 0u;                        // stride[47:32]=0, dim1_stride lo=0
    g1[7] = 0u;                        // dim1_stride (ignored for 2-D)
    asm volatile("tensor_load_to_lds %0, %1" :: "s"(g0), "s"(g1) : "memory");
}

// ---------------------------------------------------------------------------
// 9. FFN1: hidden[e,c,h] = relu(disp[e,c,:] @ w1t[e,h,:]^T + b1[e,h]).
//    Block = 32-row M-tile x all 512 cols; wave w owns cols [64w, 64w+64).
//    A staged in double-buffered LDS tiles via TDM (issued by wave 0),
//    B fragments direct from L2-resident w1t. 8 WMMAs per 32-K step.
// ---------------------------------------------------------------------------
__global__ void __launch_bounds__(256)
moe_ffn1(const _Float16* __restrict__ disp, const _Float16* __restrict__ w1t,
         const float* __restrict__ b1, _Float16* __restrict__ hidden) {
    __shared__ _Float16 abuf[2][32 * AKCH];   // 2 x 10 KB

    int wid  = threadIdx.x >> 5;
    int lane = threadIdx.x & 31;
    const int MT_PE = CAP / 32;               // 512 M-tiles per expert
    int e     = blockIdx.x / MT_PE;
    int mtile = blockIdx.x % MT_PE;
    int r0 = mtile * 32;
    int c0 = wid * 64;
    const _Float16* A = disp + (size_t)e * CAP * KPAD;
    const _Float16* B = w1t + (size_t)e * H_HID * KPAD;
    int lr = lane & 15, hb = lane >> 4, kb = hb * 8;

    const v8f vz = {0.f, 0.f, 0.f, 0.f, 0.f, 0.f, 0.f, 0.f};
    v8f acc[2][4];
#pragma unroll
    for (int mt = 0; mt < 2; ++mt)
#pragma unroll
        for (int nt = 0; nt < 4; ++nt) acc[mt][nt] = vz;

    const _Float16* atile = A + (size_t)r0 * KPAD;
    const int NCH = KPAD / AKCH;              // 5 chunks
    if (wid == 0) {
        moe_tdm_load_a((unsigned)(size_t)(&abuf[0][0]), atile);
        __builtin_amdgcn_s_wait_tensorcnt(0);
    }
    __syncthreads();

    for (int c = 0; c < NCH; ++c) {
        if (wid == 0 && c + 1 < NCH)
            moe_tdm_load_a((unsigned)(size_t)(&abuf[(c + 1) & 1][0]),
                           atile + (c + 1) * AKCH);

        const _Float16* As = &abuf[c & 1][0];
        const _Float16* Bc = B + (size_t)c * AKCH;
        for (int kk = 0; kk < AKCH; kk += 32) {
            v16h Af[2], Bf[4];
#pragma unroll
            for (int mt = 0; mt < 2; ++mt)
                Af[mt] = moe_load_frag(As + (mt * 16 + lr) * AKCH + kk + kb);
#pragma unroll
            for (int nt = 0; nt < 4; ++nt)
                Bf[nt] = moe_load_frag(Bc + (size_t)(c0 + nt * 16 + lr) * KPAD + kk + kb);
#pragma unroll
            for (int mt = 0; mt < 2; ++mt)
#pragma unroll
                for (int nt = 0; nt < 4; ++nt)
                    acc[mt][nt] = __builtin_amdgcn_wmma_f32_16x16x32_f16(
                        false, Af[mt], false, Bf[nt], (short)0, acc[mt][nt],
                        false, false);
        }

        if (wid == 0 && c + 1 < NCH)
            __builtin_amdgcn_s_wait_tensorcnt(0);
        __syncthreads();
    }

    _Float16* Hout = hidden + (size_t)e * CAP * H_HID;
#pragma unroll
    for (int mt = 0; mt < 2; ++mt) {
#pragma unroll
        for (int nt = 0; nt < 4; ++nt) {
            int col = c0 + nt * 16 + lr;
            float bias = b1[e * H_HID + col];
#pragma unroll
            for (int j = 0; j < 8; ++j) {
                int row = r0 + mt * 16 + hb * 8 + j;
                float v = acc[mt][nt][j] + bias;
                v = v > 0.f ? v : 0.f;
                Hout[(size_t)row * H_HID + col] = (_Float16)v;
            }
        }
    }
}

// ---------------------------------------------------------------------------
// 10. FFN2: eout[e,c,0:16] = hidden[e,c,:] @ w2t[e,n,:]^T + b2.
//     Wave computes a 64x16 tile; K=512 in 16 steps.
// ---------------------------------------------------------------------------
__global__ void __launch_bounds__(256)
moe_ffn2(const _Float16* __restrict__ hidden, const _Float16* __restrict__ w2t,
         const float* __restrict__ b2, float* __restrict__ eout) {
    int wid  = threadIdx.x >> 5;
    int lane = threadIdx.x & 31;
    int gw = blockIdx.x * 8 + wid;
    const int TPE = CAP / 64;  // 256 tiles per expert
    int e = gw / TPE;
    int t = gw % TPE;
    int r0 = t * 64;
    const _Float16* A = hidden + (size_t)e * CAP * H_HID;
    const _Float16* B = w2t + (size_t)e * 16 * H_HID;
    int lr = lane & 15, hb = lane >> 4, kb = hb * 8;

    const v8f vz = {0.f, 0.f, 0.f, 0.f, 0.f, 0.f, 0.f, 0.f};
    v8f acc[4];
#pragma unroll
    for (int mt = 0; mt < 4; ++mt) acc[mt] = vz;

    for (int kk = 0; kk < H_HID; kk += 32) {
        v16h Bf = moe_load_frag(B + (size_t)lr * H_HID + kk + kb);
#pragma unroll
        for (int mt = 0; mt < 4; ++mt) {
            v16h Af = moe_load_frag(A + (size_t)(r0 + mt * 16 + lr) * H_HID + kk + kb);
            acc[mt] = __builtin_amdgcn_wmma_f32_16x16x32_f16(
                false, Af, false, Bf, (short)0, acc[mt], false, false);
        }
    }

    int col = lr;
    float bias = (col < O_OUT) ? b2[e * O_OUT + col] : 0.f;
    float* Eo = eout + (size_t)e * CAP * 16;
#pragma unroll
    for (int mt = 0; mt < 4; ++mt) {
#pragma unroll
        for (int j = 0; j < 8; ++j) {
            int row = r0 + mt * 16 + hb * 8 + j;
            Eo[(size_t)row * 16 + col] = acc[mt][j] + bias;
        }
    }
}

// ---------------------------------------------------------------------------
// 11. Combine: out[s,:] = g1*eout[i1,l1,:] + g2*eout[i2,l2,:]
// ---------------------------------------------------------------------------
__global__ void __launch_bounds__(256)
moe_combine(const float* __restrict__ g1, const float* __restrict__ g2,
            const int* __restrict__ idx1, const int* __restrict__ idx2,
            const int* __restrict__ loc1, const int* __restrict__ loc2,
            const float* __restrict__ eout, float* __restrict__ out) {
    int s = blockIdx.x * 256 + threadIdx.x;
    float a = g1[s], b = g2[s];
    int l1 = loc1[s]; if (l1 < 0) l1 = 0;
    int l2 = loc2[s]; if (l2 < 0) l2 = 0;
    const float* p1 = eout + ((size_t)idx1[s] * CAP + l1) * 16;
    const float* p2 = eout + ((size_t)idx2[s] * CAP + l2) * 16;
#pragma unroll
    for (int o = 0; o < O_OUT; ++o)
        out[(size_t)s * O_OUT + o] = a * p1[o] + b * p2[o];
}

// ---------------------------------------------------------------------------
extern "C" void kernel_launch(void* const* d_in, const int* in_sizes, int n_in,
                              void* d_out, int out_size, void* d_ws, size_t ws_size,
                              hipStream_t stream) {
    (void)in_sizes; (void)n_in; (void)out_size; (void)ws_size;
    const float* x  = (const float*)d_in[0];
    const float* wg = (const float*)d_in[1];
    const float* w1 = (const float*)d_in[2];
    const float* b1 = (const float*)d_in[3];
    const float* w2 = (const float*)d_in[4];
    const float* b2 = (const float*)d_in[5];
    float* out = (float*)d_out;

    char* ws = (char*)d_ws;
    size_t off = 0;
    auto alloc = [&](size_t bytes) -> void* {
        off = (off + 255) & ~(size_t)255;
        void* p = ws + off;
        off += bytes;
        return p;
    };

    _Float16* disp   = (_Float16*)alloc((size_t)E_EXP * CAP * KPAD * 2);
    _Float16* hidden = (_Float16*)alloc((size_t)E_EXP * CAP * H_HID * 2);
    _Float16* w1t    = (_Float16*)alloc((size_t)E_EXP * H_HID * KPAD * 2);
    _Float16* w2t    = (_Float16*)alloc((size_t)E_EXP * 16 * H_HID * 2);
    float*    eout   = (float*)alloc((size_t)E_EXP * CAP * 16 * 4);
    float*    logits = (float*)alloc((size_t)S_TOK * E_EXP * 4);
    int*      idx1   = (int*)alloc((size_t)S_TOK * 4);
    int*      idx2   = (int*)alloc((size_t)S_TOK * 4);
    float*    g1     = (float*)alloc((size_t)S_TOK * 4);
    float*    g2     = (float*)alloc((size_t)S_TOK * 4);
    int*      loc1   = (int*)alloc((size_t)S_TOK * 4);
    int*      loc2   = (int*)alloc((size_t)S_TOK * 4);
    int*      cnt1   = (int*)alloc((size_t)NBLK * E_EXP * 4);
    int*      cnt2   = (int*)alloc((size_t)NBLK * E_EXP * 4);
    float*    gsum   = (float*)alloc((size_t)NBLK * E_EXP * 4);
    int*      pre1   = (int*)alloc((size_t)NBLK * E_EXP * 4);
    int*      pre2   = (int*)alloc((size_t)NBLK * E_EXP * 4);
    int*      tot1   = (int*)alloc((size_t)E_EXP * 4);

    // Routing pipeline
    moe_gate_logits<<<S_TOK / 8, 256, 0, stream>>>(x, wg, logits);
    moe_route<<<NBLK, 256, 0, stream>>>(logits, idx1, idx2, g1, g2, cnt1, cnt2, gsum);
    moe_scan<<<1, NBLK, 0, stream>>>(cnt1, cnt2, gsum, pre1, pre2, tot1,
                                     out + (size_t)S_TOK * O_OUT);
    moe_finalize<<<NBLK, 256, 0, stream>>>(idx1, idx2, pre1, pre2, tot1,
                                           g1, g2, loc1, loc2);

    // Buffer prep
    size_t disp16 = (size_t)E_EXP * CAP * KPAD * 2 / 16;
    moe_zero16<<<2048, 256, 0, stream>>>((uint4*)disp, disp16);
    moe_prep_w1<<<(E_EXP * H_HID * KPAD + 255) / 256, 256, 0, stream>>>(w1, w1t);
    moe_prep_w2<<<(E_EXP * 16 * H_HID + 255) / 256, 256, 0, stream>>>(w2, w2t);

    // Dispatch + expert FFN (WMMA, TDM-staged A) + combine
    moe_dispatch<<<S_TOK, 256, 0, stream>>>(x, idx1, idx2, loc1, loc2, disp);
    moe_ffn1<<<E_EXP * (CAP / 32), 256, 0, stream>>>(disp, w1t, b1, hidden);
    moe_ffn2<<<(E_EXP * (CAP / 64)) / 8, 256, 0, stream>>>(hidden, w2t, b2, eout);
    moe_combine<<<NBLK, 256, 0, stream>>>(g1, g2, idx1, idx2, loc1, loc2, eout, out);
}
